// PerformerClassifier_70214125355128
// MI455X (gfx1250) — matrix operand order repaired and checked by reading, not compiled, not run on hardware
//
#include <hip/hip_runtime.h>

typedef __bf16 bf16;
typedef __attribute__((ext_vector_type(16))) __bf16 v16bf;
typedef __attribute__((ext_vector_type(8)))  float  v8f;

#define DEV static __device__ __forceinline__

#define NB  110
#define NBP 112
#define TOK 16384   // B * N
#define NSEQ 4096
#define DM  256
#define DFF 1024

DEV bf16 f2bf(float f){
  unsigned u = __builtin_bit_cast(unsigned, f);
  unsigned r = u + 0x7FFFu + ((u >> 16) & 1u);
  unsigned short h = (unsigned short)(r >> 16);
  return __builtin_bit_cast(bf16, h);
}
DEV float bf2f(bf16 b){
  unsigned short h = __builtin_bit_cast(unsigned short, b);
  return __builtin_bit_cast(float, (unsigned)h << 16);
}
DEV float geluf(float v){ return 0.5f * v * (1.0f + erff(v * 0.70710678118654752f)); }
DEV void atomicMaxF(float* a, float v){
  if (v >= 0.0f) atomicMax((int*)a, __float_as_int(v));
  else           atomicMin((unsigned int*)a, (unsigned int)__float_as_int(v));
}

// ---------------- elementwise helpers ----------------
__global__ void k_f32_to_bf16(const float* __restrict__ src, bf16* __restrict__ dst, int n){
  int i = blockIdx.x * 256 + threadIdx.x;
  if (i < n) dst[i] = f2bf(src[i]);
}

__global__ void k_init_neg_inf(float* p, int n){
  int i = blockIdx.x * 64 + threadIdx.x;
  if (i < n) p[i] = -1e30f;
}

__global__ void k_zero(float* p, int n){
  int i = blockIdx.x * 256 + threadIdx.x;
  if (i < n) p[i] = 0.0f;
}

// ---------------- weight packing into WMMA B-fragment layout ----------------
// W: K x N row-major fp32.  Packed: [(nt*ktT + kt)*32 + lane]*16 bf16, where
// lane<16 : n = nbase+lane,    values[2j]=W[kbase+2j][n],  values[2j+1]=W[kbase+2j+1][n]
// lane>=16: n = nbase+lane-16, K rows kbase+16..kbase+31 likewise.
__global__ void k_pack_w(const float* __restrict__ W, bf16* __restrict__ out, int K, int N){
  int ktT = K >> 5, ntT = N >> 4;
  int total = ktT * ntT * 32;
  int gid = blockIdx.x * 256 + threadIdx.x;
  if (gid >= total) return;
  int lane = gid & 31;
  int tile = gid >> 5;
  int kt = tile % ktT;
  int nt = tile / ktT;
  int ncol = (nt << 4) + (lane & 15);
  int kbase = (kt << 5) + ((lane >> 4) << 4);
  bf16* o = out + (size_t)gid * 16;
  #pragma unroll
  for (int j = 0; j < 8; ++j){
    o[2*j]   = f2bf(W[(size_t)(kbase + 2*j    ) * N + ncol]);
    o[2*j+1] = f2bf(W[(size_t)(kbase + 2*j + 1) * N + ncol]);
  }
}

// Pack proj (NB x 32 fp32) transposed+scaled by dn into B-fragment layout:
// Wt[k][n] = proj[n][k]*dn, K=32 (ktT=1), N=112 (ntT=7), zeros for n>=NB.
__global__ void k_pack_proj(const float* __restrict__ proj, bf16* __restrict__ PP){
  const int gid = threadIdx.x;       // 256 threads, 1 block; 224 active
  if (gid >= 7*32) return;
  const int lane = gid & 31;
  const int nt = gid >> 5;           // 0..6
  const int ncol = (nt << 4) + (lane & 15);
  const int kbase = (lane >> 4) << 4;
  const float dn = 0.42044820762685725f;   // 32^-0.25
  bf16* o = PP + (size_t)gid * 16;
  #pragma unroll
  for (int j = 0; j < 8; ++j){
    const int k0 = kbase + 2*j;
    float v0 = (ncol < NB) ? proj[ncol*32 + k0    ] * dn : 0.0f;
    float v1 = (ncol < NB) ? proj[ncol*32 + k0 + 1] * dn : 0.0f;
    o[2*j]   = f2bf(v0);
    o[2*j+1] = f2bf(v1);
  }
}

// ---------------- WMMA bf16 GEMM, fused epilogue ----------------
// EPI bits: 1=bias, 2=residual add, 4=gelu, 8=store fp32, 16=store bf16
template<int EPI>
__global__ __launch_bounds__(256, 2)
void k_gemm_bf16(const bf16* __restrict__ A, const bf16* __restrict__ Bp,
                 const float* __restrict__ bias, const float* __restrict__ res,
                 float* __restrict__ Cf, bf16* __restrict__ Cb,
                 int M, int N, int K)
{
  const int lane = threadIdx.x & 31;
  const int wave = threadIdx.x >> 5;
  const int wm = wave & 3, wn = wave >> 2;          // 4x2 waves => 128x64 block tile
  const int m0 = blockIdx.x * 128 + wm * 32;
  const int n0 = blockIdx.y * 64  + wn * 32;
  const int l16 = lane & 15, lhi = lane >> 4;
  const int ktT = K >> 5;
  (void)M;

  v8f acc[2][2];
  #pragma unroll
  for (int i = 0; i < 2; ++i)
    #pragma unroll
    for (int j = 0; j < 2; ++j)
      #pragma unroll
      for (int r = 0; r < 8; ++r) acc[i][j][r] = 0.0f;

  union U { v16bf v; uint4 q[2]; };

  const bf16* pa0 = A + (size_t)(m0 + l16)      * K + lhi * 16;
  const bf16* pa1 = A + (size_t)(m0 + 16 + l16) * K + lhi * 16;
  const int nt0 = n0 >> 4;
  const bf16* pb0 = Bp + ((size_t)(nt0    ) * ktT * 32 + lane) * 16;
  const bf16* pb1 = Bp + ((size_t)(nt0 + 1) * ktT * 32 + lane) * 16;

  for (int kt = 0; kt < ktT; ++kt){
    const int k0 = kt << 5;
    U a0, a1, b0, b1;
    a0.q[0] = *(const uint4*)(pa0 + k0);
    a0.q[1] = *(const uint4*)(pa0 + k0 + 8);
    a1.q[0] = *(const uint4*)(pa1 + k0);
    a1.q[1] = *(const uint4*)(pa1 + k0 + 8);
    b0.q[0] = *(const uint4*)(pb0 + (size_t)kt * 512);
    b0.q[1] = *(const uint4*)(pb0 + (size_t)kt * 512 + 8);
    b1.q[0] = *(const uint4*)(pb1 + (size_t)kt * 512);
    b1.q[1] = *(const uint4*)(pb1 + (size_t)kt * 512 + 8);

    __builtin_prefetch(pa0 + k0 + 32, 0, 1);   // next K tile of streaming A
    __builtin_prefetch(pa1 + k0 + 32, 0, 1);

    acc[0][0] = __builtin_amdgcn_wmma_f32_16x16x32_bf16(false, a0.v, false, b0.v, (short)0, acc[0][0], false, false);
    acc[0][1] = __builtin_amdgcn_wmma_f32_16x16x32_bf16(false, a0.v, false, b1.v, (short)0, acc[0][1], false, false);
    acc[1][0] = __builtin_amdgcn_wmma_f32_16x16x32_bf16(false, a1.v, false, b0.v, (short)0, acc[1][0], false, false);
    acc[1][1] = __builtin_amdgcn_wmma_f32_16x16x32_bf16(false, a1.v, false, b1.v, (short)0, acc[1][1], false, false);
  }

  #pragma unroll
  for (int j = 0; j < 2; ++j){
    const int n = n0 + j * 16 + l16;
    const float bv = (EPI & 1) ? bias[n] : 0.0f;
    #pragma unroll
    for (int i = 0; i < 2; ++i){
      #pragma unroll
      for (int r = 0; r < 8; ++r){
        const int m = m0 + i * 16 + lhi * 8 + r;
        const size_t idx = (size_t)m * N + n;
        float vv = acc[i][j][r] + bv;
        if (EPI & 2)  vv += res[idx];
        if (EPI & 4)  vv = geluf(vv);
        if (EPI & 8)  Cf[idx] = vv;
        if (EPI & 16) Cb[idx] = f2bf(vv);
      }
    }
  }
}

// ---------------- Performer feature map, fully fused on WMMA ----------------
// Per wave: 16 tokens x 112 features, 7 WMMAs (K=32, dn folded into PP).
// MODE 0: q path  -> per-row max (in-register, shfl) + exp -> write bf16 qp
// MODE 1: k pass1 -> per-row max -> atomicMax into KM[bh] (no stores)
// MODE 2: k pass2 -> recompute dd, exp with global max KM[bh] -> write bf16 kp
template<int MODE>
__global__ __launch_bounds__(256, 1)
void k_featmap(const bf16* __restrict__ Xb, const bf16* __restrict__ PP,
               float* __restrict__ KM, bf16* __restrict__ OP)
{
  const int lane = threadIdx.x & 31;
  const int wave = threadIdx.x >> 5;
  const int m0 = blockIdx.x * 128 + wave * 16;   // token base for this wave
  const int h  = blockIdx.z;
  const int l16 = lane & 15, lhi = lane >> 4;

  union U { v16bf v; uint4 q[2]; };
  U a, bfr[7];
  const bf16* pa = Xb + (size_t)(m0 + l16) * DM + h * 32 + lhi * 16;
  a.q[0] = *(const uint4*)(pa);
  a.q[1] = *(const uint4*)(pa + 8);
  #pragma unroll
  for (int j = 0; j < 7; ++j){
    const bf16* pb = PP + ((size_t)j * 32 + lane) * 16;
    bfr[j].q[0] = *(const uint4*)(pb);
    bfr[j].q[1] = *(const uint4*)(pb + 8);
  }

  v8f acc[7];
  #pragma unroll
  for (int j = 0; j < 7; ++j){
    #pragma unroll
    for (int r = 0; r < 8; ++r) acc[j][r] = 0.0f;
    acc[j] = __builtin_amdgcn_wmma_f32_16x16x32_bf16(false, a.v, false, bfr[j].v, (short)0, acc[j], false, false);
  }

  // per-row max over valid features (C-row of lane = m0 + lhi*8 + r)
  float mxr[8];
  #pragma unroll
  for (int r = 0; r < 8; ++r){
    float mx = -1e30f;
    #pragma unroll
    for (int j = 0; j < 7; ++j){
      float v = acc[j][r];
      if (j == 6 && l16 >= 14) v = -1e30f;   // features 110,111 are padding
      mx = fmaxf(mx, v);
    }
    #pragma unroll
    for (int o = 8; o > 0; o >>= 1) mx = fmaxf(mx, __shfl_xor(mx, o, 32));
    mxr[r] = mx;
  }

  const int bh = ((m0 >> 12) * 8 + h);   // batch constant per block (128 | 4096)

  if (MODE == 1){
    if (l16 == 0){
      float m8 = -1e30f;
      #pragma unroll
      for (int r = 0; r < 8; ++r) m8 = fmaxf(m8, mxr[r]);
      atomicMaxF(&KM[bh], m8);
    }
    return;
  }

  // sum of squares of this lane's half of A-row (row m0+l16); combine halves
  float sqp = 0.0f;
  #pragma unroll
  for (int e = 0; e < 16; ++e){ float t = bf2f(a.v[e]); sqp += t * t; }
  const float sqrow = sqp + __shfl_xor(sqp, 16, 32);

  const float gmx = (MODE == 2) ? KM[bh] : 0.0f;
  const size_t outbase = (size_t)bh * NSEQ;
  const int n_in_b = m0 & (NSEQ-1);
  const float ratio = 0.09534625892455924f;              // 110^-0.5
  #pragma unroll
  for (int r = 0; r < 8; ++r){
    const int row = lhi*8 + r;
    const float diag = 0.5f * 0.17677669529663687f * __shfl(sqrow, row, 32);
    const float mx = (MODE == 0) ? mxr[r] : gmx;
    bf16* orow = OP + (outbase + n_in_b + row) * NBP;
    #pragma unroll
    for (int j = 0; j < 7; ++j){
      const int feat = j*16 + l16;
      float e = ratio * (__expf(acc[j][r] - diag - mx) + 1e-4f);
      if (feat >= NB) e = 0.0f;
      orow[feat] = f2bf(e);
    }
  }
}

// ---------------- LayerNorm (one wave per 256-wide row), bf16 out ----------------
__global__ __launch_bounds__(256)
void k_layernorm_bf16(const float* __restrict__ X, const float* __restrict__ g,
                      const float* __restrict__ b, bf16* __restrict__ Y, int rows)
{
  const int lane = threadIdx.x & 31;
  const int row = blockIdx.x * 8 + (threadIdx.x >> 5);
  if (row >= rows) return;
  const float* xr = X + (size_t)row * DM;
  float v[8]; float s = 0.0f, sq = 0.0f;
  #pragma unroll
  for (int e = 0; e < 8; ++e){ float t = xr[lane + 32*e]; v[e] = t; s += t; sq += t*t; }
  #pragma unroll
  for (int o = 16; o > 0; o >>= 1){ s += __shfl_xor(s, o, 32); sq += __shfl_xor(sq, o, 32); }
  const float mu = s * (1.0f/256.0f);
  const float var = sq * (1.0f/256.0f) - mu * mu;
  const float rs = rsqrtf(var + 1e-5f);
  bf16* yr = Y + (size_t)row * DM;
  #pragma unroll
  for (int e = 0; e < 8; ++e){ int c = lane + 32*e; yr[c] = f2bf((v[e]-mu)*rs*g[c] + b[c]); }
}

// ---------------- k_sum over sequence (kp bf16), chunked + atomic ----------------
#define KCH (NSEQ/8)
__global__ __launch_bounds__(128)
void k_ksum(const bf16* __restrict__ KP, float* __restrict__ KS){
  const int bh = blockIdx.x, m = threadIdx.x, c = blockIdx.y;
  if (m >= NBP) return;
  const bf16* base = KP + ((size_t)bh * NSEQ + (size_t)c * KCH) * NBP + m;
  float s = 0.0f;
  for (int n = 0; n < KCH; ++n) s += bf2f(base[(size_t)n * NBP]);
  atomicAdd(&KS[bh * NBP + m], s);
}

// ---------------- ctx[m][d] += sum_n kp[n][m] * v[n][d], chunked + atomic ----------------
__global__ __launch_bounds__(256)
void k_ctx(const bf16* __restrict__ KP, const float* __restrict__ V,
           float* __restrict__ CTX)
{
  __shared__ float skp[32 * NBP];
  __shared__ float sv[32 * 32];
  const int bh = blockIdx.x;
  const int b = bh >> 3, hh = bh & 7;
  const int nbase = blockIdx.y * KCH;
  const int t = threadIdx.x;
  const int d = t & 31, mb = t >> 5;   // mb in 0..7
  float acc[14];
  #pragma unroll
  for (int j = 0; j < 14; ++j) acc[j] = 0.0f;
  const bf16* kpb = KP + ((size_t)bh * NSEQ + nbase) * NBP;
  const float* vb  = V  + ((size_t)b * NSEQ + nbase) * DM + hh * 32;
  for (int n0 = 0; n0 < KCH; n0 += 32){
    for (int i = t; i < 32*NBP; i += 256) skp[i] = bf2f(kpb[(size_t)n0 * NBP + i]);
    for (int i = t; i < 32*32;  i += 256) sv[i]  = vb[(size_t)(n0 + (i>>5)) * DM + (i & 31)];
    __syncthreads();
    for (int nn = 0; nn < 32; ++nn){
      const float vv = sv[nn*32 + d];
      #pragma unroll
      for (int j = 0; j < 14; ++j) acc[j] += skp[nn*NBP + mb + 8*j] * vv;
    }
    __syncthreads();
  }
  float* cb = CTX + (size_t)bh * NBP * 32;
  #pragma unroll
  for (int j = 0; j < 14; ++j) atomicAdd(&cb[(mb + 8*j) * 32 + d], acc[j]);
}

// ---------------- out[n][d] = (qp[n]·ctx[:,d]) / (qp[n]·k_sum) ----------------
__global__ __launch_bounds__(128)
void k_attn_out(const bf16* __restrict__ QP, const float* __restrict__ CTX,
                const float* __restrict__ KS, float* __restrict__ Ao)
{
  __shared__ float sctx[NBP * 32];
  __shared__ float sks[NBP];
  const int hh = blockIdx.y;
  const int t = threadIdx.x;
  const int token = blockIdx.x * 128 + t;
  const int b = token >> 12, n = token & (NSEQ-1);
  const int bh = b*8 + hh;
  for (int i = t; i < NBP*32; i += 128) sctx[i] = CTX[(size_t)bh * NBP * 32 + i];
  if (t < NBP) sks[t] = KS[bh * NBP + t];
  __syncthreads();
  const bf16* qr = QP + ((size_t)bh * NSEQ + n) * NBP;
  float acc[32];
  #pragma unroll
  for (int d = 0; d < 32; ++d) acc[d] = 0.0f;
  float den = 0.0f;
  for (int m = 0; m < NBP; ++m){
    const float qv = bf2f(qr[m]);
    den += qv * sks[m];
    #pragma unroll
    for (int d = 0; d < 32; ++d) acc[d] += qv * sctx[m*32 + d];
  }
  const float di = 1.0f / den;
  float* ar = Ao + (size_t)token * DM + hh * 32;
  #pragma unroll
  for (int d = 0; d < 32; ++d) ar[d] = acc[d] * di;
}

// ---------------- mean pool over sequence (chunked + atomic) ----------------
#define MCH (NSEQ/16)
__global__ void k_mean_part(const float* __restrict__ H, float* __restrict__ P){
  const int b = blockIdx.x, c = threadIdx.x;
  const float* base = H + ((size_t)b * NSEQ + (size_t)blockIdx.y * MCH) * DM + c;
  float s = 0.0f;
  for (int n = 0; n < MCH; ++n) s += base[(size_t)n * DM];
  atomicAdd(&P[b * DM + c], s);
}

// ---------------- classifier head (tiny); P holds sums, scale by 1/NSEQ ----------------
__global__ __launch_bounds__(256)
void k_head(const float* __restrict__ P, const float* __restrict__ g,
            const float* __restrict__ bb, const float* __restrict__ wc1,
            const float* __restrict__ bc1, const float* __restrict__ wc2,
            const float* __restrict__ bc2, float* __restrict__ out)
{
  __shared__ float red[256];
  __shared__ float s1[256];
  __shared__ float s2[256];
  const int b = blockIdx.x, t = threadIdx.x;
  float x = P[b * DM + t] * (1.0f / NSEQ);
  red[t] = x; __syncthreads();
  for (int s = 128; s > 0; s >>= 1){ if (t < s) red[t] += red[t + s]; __syncthreads(); }
  const float mu = red[0] * (1.0f/256.0f); __syncthreads();
  const float dx = x - mu;
  red[t] = dx * dx; __syncthreads();
  for (int s = 128; s > 0; s >>= 1){ if (t < s) red[t] += red[t + s]; __syncthreads(); }
  const float var = red[0] * (1.0f/256.0f);
  float z = fmaxf(dx * rsqrtf(var + 1e-5f) * g[t] + bb[t], 0.0f);
  s1[t] = z; __syncthreads();
  float z2 = bc1[t];
  for (int k2 = 0; k2 < 256; ++k2) z2 += s1[k2] * wc1[k2*256 + t];
  z2 = fmaxf(z2, 0.0f);
  s2[t] = z2; __syncthreads();
  if (t < 10){
    float o = bc2[t];
    for (int k2 = 0; k2 < 256; ++k2) o += s2[k2] * wc2[k2*10 + t];
    out[b*10 + t] = o;
  }
}

// ---------------- workspace sizes ----------------
static constexpr size_t SZ_H   = (size_t)TOK * DM * 4;
static constexpr size_t SZ_ACT = (size_t)TOK * DM * 2;
static constexpr size_t SZ_G   = (size_t)TOK * DFF * 2;
static constexpr size_t SZ_V   = (size_t)TOK * DM * 4;
static constexpr size_t SZ_QPB = (size_t)32 * NSEQ * NBP * 2;   // bf16 qp/kp
static constexpr size_t SZ_WP  = (size_t)DM * DFF * 2;
static constexpr size_t SZ_PP  = (size_t)7 * 32 * 16 * 2;
static constexpr size_t SZ_CTX = (size_t)32 * NBP * 32 * 4;
static constexpr size_t SZ_KS  = (size_t)32 * NBP * 4;
static constexpr size_t SZ_KM  = 32 * 4;
static constexpr size_t SZ_P   = 4 * DM * 4;

extern "C" void kernel_launch(void* const* d_in, const int* in_sizes, int n_in,
                              void* d_out, int out_size, void* d_ws, size_t ws_size,
                              hipStream_t stream)
{
  (void)in_sizes; (void)n_in; (void)out_size; (void)ws_size;
  const float* x    = (const float*)d_in[0];
  const float* w_in = (const float*)d_in[2];
  const float* b_in = (const float*)d_in[3];
  const float* ln1g = (const float*)d_in[4];
  const float* ln1b = (const float*)d_in[5];
  const float* wq   = (const float*)d_in[6];
  const float* wk   = (const float*)d_in[7];
  const float* wv   = (const float*)d_in[8];
  const float* wo   = (const float*)d_in[9];
  const float* bo   = (const float*)d_in[10];
  const float* proj = (const float*)d_in[11];
  const float* ln2g = (const float*)d_in[12];
  const float* ln2b = (const float*)d_in[13];
  const float* w1   = (const float*)d_in[14];
  const float* b1   = (const float*)d_in[15];
  const float* w2   = (const float*)d_in[16];
  const float* b2   = (const float*)d_in[17];
  const float* clsg = (const float*)d_in[18];
  const float* clsb = (const float*)d_in[19];
  const float* wc1  = (const float*)d_in[20];
  const float* bc1  = (const float*)d_in[21];
  const float* wc2  = (const float*)d_in[22];
  const float* bc2  = (const float*)d_in[23];
  float* out = (float*)d_out;

  char* ws = (char*)d_ws;
  size_t off = 0;
  auto take = [&](size_t bytes)->char*{
    char* p = ws + off; off += (bytes + 255) & ~(size_t)255; return p;
  };
  float* H   = (float*)take(SZ_H);
  bf16*  ACT = (bf16*)take(SZ_ACT);
  bf16*  GB  = (bf16*)take(SZ_G);
  bf16*  Qb  = (bf16*)take(SZ_ACT);
  bf16*  Kb  = (bf16*)take(SZ_ACT);
  float* V   = (float*)take(SZ_V);
  bf16*  QP  = (bf16*)take(SZ_QPB);
  bf16*  KP  = (bf16*)take(SZ_QPB);
  float* Ab  = (float*)take(SZ_V);
  bf16*  WP  = (bf16*)take(SZ_WP);
  bf16*  PP  = (bf16*)take(SZ_PP);
  float* CTX = (float*)take(SZ_CTX);
  float* KS  = (float*)take(SZ_KS);
  float* KM  = (float*)take(SZ_KM);
  float* P   = (float*)take(SZ_P);

  // input projection: h = x @ w_in + b_in   (M=16384, K=128, N=256)
  k_f32_to_bf16<<<(TOK*128)/256, 256, 0, stream>>>(x, ACT, TOK*128);
  k_pack_w<<<(4*16*32 + 255)/256, 256, 0, stream>>>(w_in, WP, 128, DM);
  k_gemm_bf16<9><<<dim3(TOK/128, DM/64), 256, 0, stream>>>(ACT, WP, b_in, nullptr, H, nullptr, TOK, DM, 128);

  for (int i = 0; i < 6; ++i){
    const float* wq_i = wq + (size_t)i * DM * DM;
    const float* wk_i = wk + (size_t)i * DM * DM;
    const float* wv_i = wv + (size_t)i * DM * DM;
    const float* wo_i = wo + (size_t)i * DM * DM;
    const float* w1_i = w1 + (size_t)i * DM * DFF;
    const float* w2_i = w2 + (size_t)i * DFF * DM;
    const float* pr_i = proj + (size_t)i * NB * 32;

    k_layernorm_bf16<<<TOK/8, 256, 0, stream>>>(H, ln1g + i*DM, ln1b + i*DM, ACT, TOK);
    k_pack_w<<<16, 256, 0, stream>>>(wq_i, WP, DM, DM);
    k_gemm_bf16<16><<<dim3(TOK/128, DM/64), 256, 0, stream>>>(ACT, WP, nullptr, nullptr, nullptr, Qb, TOK, DM, DM);
    k_pack_w<<<16, 256, 0, stream>>>(wk_i, WP, DM, DM);
    k_gemm_bf16<16><<<dim3(TOK/128, DM/64), 256, 0, stream>>>(ACT, WP, nullptr, nullptr, nullptr, Kb, TOK, DM, DM);
    k_pack_w<<<16, 256, 0, stream>>>(wv_i, WP, DM, DM);
    k_gemm_bf16<8><<<dim3(TOK/128, DM/64), 256, 0, stream>>>(ACT, WP, nullptr, nullptr, V, nullptr, TOK, DM, DM);

    // Performer feature maps: fused WMMA dd + max + exp
    k_pack_proj<<<1, 256, 0, stream>>>(pr_i, PP);
    k_featmap<0><<<dim3(TOK/128, 1, 8), 256, 0, stream>>>(Qb, PP, nullptr, QP);
    k_init_neg_inf<<<1, 64, 0, stream>>>(KM, 32);
    k_featmap<1><<<dim3(TOK/128, 1, 8), 256, 0, stream>>>(Kb, PP, KM, nullptr);
    k_featmap<2><<<dim3(TOK/128, 1, 8), 256, 0, stream>>>(Kb, PP, KM, KP);

    // linear attention: k_sum, ctx (chunked, atomic accumulate), out
    k_zero<<<(32*NBP + 255)/256, 256, 0, stream>>>(KS, 32*NBP);
    k_zero<<<(32*NBP*32 + 255)/256, 256, 0, stream>>>(CTX, 32*NBP*32);
    k_ksum<<<dim3(32, 8), 128, 0, stream>>>(KP, KS);
    k_ctx<<<dim3(32, 8), 256, 0, stream>>>(KP, V, CTX);
    k_attn_out<<<dim3(TOK/128, 8), 128, 0, stream>>>(QP, CTX, KS, Ab);

    k_f32_to_bf16<<<(TOK*DM)/256, 256, 0, stream>>>(Ab, ACT, TOK*DM);
    k_pack_w<<<16, 256, 0, stream>>>(wo_i, WP, DM, DM);
    k_gemm_bf16<11><<<dim3(TOK/128, DM/64), 256, 0, stream>>>(ACT, WP, bo + i*DM, H, H, nullptr, TOK, DM, DM);

    k_layernorm_bf16<<<TOK/8, 256, 0, stream>>>(H, ln2g + i*DM, ln2b + i*DM, ACT, TOK);
    k_pack_w<<<64, 256, 0, stream>>>(w1_i, WP, DM, DFF);
    k_gemm_bf16<21><<<dim3(TOK/128, DFF/64), 256, 0, stream>>>(ACT, WP, b1 + i*DFF, nullptr, nullptr, GB, TOK, DFF, DM);
    k_pack_w<<<64, 256, 0, stream>>>(w2_i, WP, DFF, DM);
    k_gemm_bf16<11><<<dim3(TOK/128, DM/64), 256, 0, stream>>>(GB, WP, b2 + i*DM, H, H, nullptr, TOK, DM, DFF);
  }

  k_zero<<<4, 256, 0, stream>>>(P, 4*DM);
  k_mean_part<<<dim3(4, 16), 256, 0, stream>>>(H, P);
  k_head<<<4, 256, 0, stream>>>(P, clsg, clsb, wc1, bc1, wc2, bc2, out);
}
